// MeshNet_29137058136342
// MI455X (gfx1250) — compile-verified, matrix operand order checked by reading, not run on hardware
//
#include <hip/hip_runtime.h>
#include <hip/hip_bf16.h>

// ---------------- problem constants ----------------
#define BS    8
#define NE    4800
#define NN    4981
#define CIN   3
#define HDIM  128
#define DBLK  6
#define NEDGE (BS * 19200)      // 153600
#define NNODE (BS * NE)         // 38400

typedef __attribute__((ext_vector_type(16))) _Float16 v16h;
typedef __attribute__((ext_vector_type(8)))  _Float16 v8h;
typedef __attribute__((ext_vector_type(8)))  float    v8f;
typedef __attribute__((ext_vector_type(4)))  float    v4f;
typedef __attribute__((ext_vector_type(4)))  unsigned int v4u;
typedef __attribute__((ext_vector_type(8)))  int      v8i;
typedef __attribute__((ext_vector_type(4)))  int      v4i;

#if defined(__has_builtin)
#if __has_builtin(__builtin_amdgcn_tensor_load_to_lds)
#define HAVE_TDM 1
#endif
#endif
#if __has_include(<hip/amd_detail/amd_gfx1250_TDM.h>)
#define TDM_6ARG 1   // amdgpu-toolchain (clang-23 / therock-10.0 headers): 6-arg builtin
#endif

__device__ __forceinline__ float lrelu(float x) { return x >= 0.f ? x : 0.2f * x; }

// ============================================================
// WMMA GEMM: C[M x 128] = act(A[M x K] * W^T + bias)
// A fp16 row-major (lda = K), Wt fp16 stored [128][K] (transposed).
// The ENTIRE weight matrix (<= 64KB) is staged into LDS once via the
// Tensor Data Mover (1-D descriptor, 8-byte elements) issued by wave 0,
// completion enforced with s_wait_tensorcnt + workgroup barrier. Then the
// K-loop runs barrier-free: global A-fragment loads overlap ds_load->wmma.
// Block: 256 threads = 8 waves; wave w computes rows [blk*128 + w*16, +16),
// all 128 cols (8 WMMA n-tiles, 8 x v8f accumulators).
// M must be a multiple of 128 (true for all launches here).
// ============================================================
template<int KTILES, bool LRELU_ACT, bool HAS_BIAS, bool OUTF16>
__global__ __launch_bounds__(256)
void wmma_mlp_layer(const _Float16* __restrict__ A,
                    const _Float16* __restrict__ Wt,
                    const float* __restrict__ bias,
                    float* __restrict__ Cf,
                    _Float16* __restrict__ Ch)
{
    constexpr int K = KTILES * 32;
    __shared__ _Float16 Bs[128 * K];    // full transposed weights, [n][k]

    const int tid  = threadIdx.x;
    const int wave = tid >> 5;
    const int lane = tid & 31;
    const int hi16 = lane >> 4;         // 0: lanes 0-15, 1: lanes 16-31
    const int l15  = lane & 15;
    const int rowB = blockIdx.x * 128 + wave * 16;
    const int arow = rowB + l15;

#ifdef HAVE_TDM
    // ---- TDM: one DMA for the whole 128 x K fp16 weight block ----
    if (wave == 0) {
        const unsigned long long ga = (unsigned long long)(uintptr_t)Wt;
        const unsigned int galo = (unsigned int)ga;
        const unsigned int gahi = (unsigned int)(ga >> 32);
        const unsigned int ldsb = (unsigned int)(uintptr_t)(void*)Bs;   // LDS byte offset in [31:0]
        const unsigned int elems = (unsigned int)(128 * K / 4);         // 8-byte units
        // D# group 0: count=1 | lds_addr | global_addr[56:0] | type=2
        v4u g0 = { 1u, ldsb, galo, (gahi & 0x01FFFFFFu) | (2u << 30) };
        // D# group 1: data_size=3 (8B); tensor_dim0=tile_dim0=elems;
        // tensor_dim1=tile_dim1=1; tensor_dim0_stride=elems; no mask/pad/iterate.
        v8i g1 = { (int)(3u << 16),
                   (int)((elems & 0xFFFFu) << 16),
                   (int)((elems >> 16) | (1u << 16)),
                   (int)((elems & 0xFFFFu) << 16),
                   1, (int)elems, 0, 0 };
        v4i gz = { 0, 0, 0, 0 };
#ifdef TDM_6ARG
        v8i gz8 = { 0, 0, 0, 0, 0, 0, 0, 0 };
        __builtin_amdgcn_tensor_load_to_lds(g0, g1, gz, gz, gz8, 0);
#else
        __builtin_amdgcn_tensor_load_to_lds(g0, g1, gz, gz, 0);
#endif
        __builtin_amdgcn_s_wait_tensorcnt(0);
    }
    __syncthreads();
#else
    // fallback: cooperative staged copy (coalesced 16B), one barrier total
#pragma unroll 1
    for (int i = tid * 8; i < 128 * K; i += 256 * 8)
        *(v8h*)(Bs + i) = *(const v8h*)(Wt + i);
    __syncthreads();
#endif

    v8f acc[8];
#pragma unroll
    for (int j = 0; j < 8; ++j)
#pragma unroll
        for (int r = 0; r < 8; ++r) acc[j][r] = 0.f;

#pragma unroll 1
    for (int kc = 0; kc < KTILES; ++kc) {
        // A fragment: per ISA 16-bit A layout (lanes 0-15: K 0-7 & 16-23,
        // lanes 16-31: K 8-15 & 24-31), two 16B global loads per lane.
        const _Float16* ap = A + (size_t)arow * K + kc * 32 + hi16 * 8;
        if (kc + 1 < KTILES) __builtin_prefetch(ap + 32, 0, 0);
        v8h alo = *(const v8h*)ap;
        v8h ahi = *(const v8h*)(ap + 16);
        v16h a;
#pragma unroll
        for (int i = 0; i < 8; ++i) { a[i] = alo[i]; a[8 + i] = ahi[i]; }

#pragma unroll
        for (int j = 0; j < 8; ++j) {
            const _Float16* bp = Bs + (size_t)(j * 16 + l15) * K + kc * 32 + hi16 * 8;
            v8h blo = *(const v8h*)bp;
            v8h bhi = *(const v8h*)(bp + 16);
            v16h b;
#pragma unroll
            for (int i = 0; i < 8; ++i) { b[i] = blo[i]; b[8 + i] = bhi[i]; }
            acc[j] = __builtin_amdgcn_wmma_f32_16x16x32_f16(
                false, a, false, b, (short)0, acc[j], false, false);
        }
    }

    // epilogue: C layout -> VGPR r: lanes 0-15 row r, lanes 16-31 row r+8.
    // Unconditional stores (M multiple of 128), bias resolved at compile time.
#pragma unroll
    for (int j = 0; j < 8; ++j) {
        const int col = j * 16 + l15;
        const float bv = HAS_BIAS ? bias[col] : 0.f;
#pragma unroll
        for (int r = 0; r < 8; ++r) {
            const int row = rowB + r + hi16 * 8;
            float v = acc[j][r] + bv;
            if (LRELU_ACT) v = lrelu(v);
            if (OUTF16) Ch[(size_t)row * 128 + col] = (_Float16)v;
            else        Cf[(size_t)row * 128 + col] = v;
        }
    }
}

// ============================================================
// LayerNorm over rows of 128 floats; one wave per row (wave32),
// shfl_xor reduction; in-place f32 plus optional fp16 mirror.
// ============================================================
__global__ __launch_bounds__(256)
void layernorm128(float* __restrict__ X, _Float16* __restrict__ X16,
                  const float* __restrict__ g, const float* __restrict__ b, int M)
{
    const int wave = threadIdx.x >> 5, lane = threadIdx.x & 31;
    const int row = blockIdx.x * 8 + wave;
    if (row >= M) return;
    float* xp = X + (size_t)row * 128 + lane * 4;
    v4f x = *(v4f*)xp;
    float s  = x[0] + x[1] + x[2] + x[3];
    float sq = x[0]*x[0] + x[1]*x[1] + x[2]*x[2] + x[3]*x[3];
#pragma unroll
    for (int m = 16; m >= 1; m >>= 1) {
        s  += __shfl_xor(s,  m, 32);
        sq += __shfl_xor(sq, m, 32);
    }
    const float mean = s * (1.f / 128.f);
    const float var  = sq * (1.f / 128.f) - mean * mean;
    const float rstd = rsqrtf(var + 1e-5f);
#pragma unroll
    for (int i = 0; i < 4; ++i) {
        const int c = lane * 4 + i;
        const float y = (x[i] - mean) * rstd * g[c] + b[c];
        xp[i] = y;
        if (X16) X16[(size_t)row * 128 + c] = (_Float16)y;
    }
}

// ---- weight convert: fp32 W[d][K][N] -> fp16 Wt[d][N][Kpad] (zero-padded K) ----
__global__ void convert_wt(const float* __restrict__ W, _Float16* __restrict__ Wt,
                           int K, int Kpad, int NC, long long total)
{
    long long t = blockIdx.x * (long long)blockDim.x + threadIdx.x;
    if (t >= total) return;
    const long long per = (long long)NC * Kpad;
    const int d = (int)(t / per);
    const long long rem = t % per;
    const int n = (int)(rem / Kpad);
    const int k = (int)(rem % Kpad);
    Wt[(size_t)d * per + (size_t)n * Kpad + k] =
        (k < K) ? (_Float16)W[((size_t)d * K + k) * NC + n] : (_Float16)0.f;
}

__global__ void fill_zero_f32(float* __restrict__ p, long long n)
{
    long long t = blockIdx.x * (long long)blockDim.x + threadIdx.x;
    if (t < n) p[t] = 0.f;
}

// ---- encoder per-channel conv (4 mesh nodes -> 1) + LN(axis size 1) ----
// writes fp16 zero-padded rows of width 32 for the WMMA encoder MLP
__global__ void encoder_conv(const float* __restrict__ x, const int* __restrict__ ec,
                             const float* __restrict__ cw1, const float* __restrict__ cb1,
                             const float* __restrict__ cw2, const float* __restrict__ cb2,
                             const float* __restrict__ cw3, const float* __restrict__ clng,
                             const float* __restrict__ clnb,
                             _Float16* __restrict__ xpad, int ntot)
{
    int t = blockIdx.x * blockDim.x + threadIdx.x;
    if (t >= ntot) return;
    const int b = t / NE, e = t % NE;
    float out[CIN];
#pragma unroll
    for (int c = 0; c < CIN; ++c) {
        float acc = cb1[c];
#pragma unroll
        for (int i = 0; i < 4; ++i) {
            const int node = ec[e * 4 + i];
            acc += x[((size_t)b * NN + node) * CIN + c] * cw1[c * 4 + i];
        }
        acc = lrelu(acc);
        acc = lrelu(acc * cw2[c] + cb2[c]);
        acc = acc * cw3[c];
        // LayerNorm over size-1 axis: mean == value, var == 0
        out[c] = (acc - acc) * rsqrtf(1e-5f) * clng[c] + clnb[c];
    }
    _Float16* xp = xpad + (size_t)t * 32;
#pragma unroll
    for (int c = 0; c < 32; ++c) xp[c] = (_Float16)(c < CIN ? out[c] : 0.f);
}

// ---- edge gather: msg[e] = [h16[dest], h16[src]]  (16B vector copies) ----
__global__ void gather_msg(const _Float16* __restrict__ h16, const int* __restrict__ ei,
                           _Float16* __restrict__ msg, int nedge)
{
    long long t = blockIdx.x * (long long)blockDim.x + threadIdx.x;
    if (t >= (long long)nedge * 32) return;
    const int e = (int)(t >> 5), j = (int)(t & 31);
    const int part = (j < 16) ? 0 : 1;                      // 0 -> dest, 1 -> src
    const int rown = part ? ei[e] : ei[nedge + e];          // src row / dest row
    const int cj = (j & 15) * 8;
    const v8h v = *(const v8h*)(h16 + (size_t)rown * 128 + cj);
    *(v8h*)(msg + (size_t)e * 256 + part * 128 + cj) = v;
}

// ---- segment-sum via f32 global atomics ----
__global__ void scatter_add_edges(const float* __restrict__ m, const int* __restrict__ ei,
                                  float* __restrict__ aggr, int nedge)
{
    long long t = blockIdx.x * (long long)blockDim.x + threadIdx.x;
    if (t >= (long long)nedge * 128) return;
    const int e = (int)(t >> 7), c = (int)(t & 127);
    const int d = ei[nedge + e];
    atomicAdd(aggr + (size_t)d * 128 + c, m[t]);
}

// ---- node concat: nin[i] = [h16[i], fp16(aggr[i])] ----
__global__ void concat_node(const _Float16* __restrict__ h16, const float* __restrict__ aggr,
                            _Float16* __restrict__ nin, int n)
{
    long long t = blockIdx.x * (long long)blockDim.x + threadIdx.x;
    if (t >= (long long)n * 32) return;
    const int i = (int)(t >> 5), j = (int)(t & 31);
    if (j < 16) {
        const int cj = j * 8;
        *(v8h*)(nin + (size_t)i * 256 + cj) = *(const v8h*)(h16 + (size_t)i * 128 + cj);
    } else {
        const int cj = (j - 16) * 8;
        const float* a = aggr + (size_t)i * 128 + cj;
        v8h v;
#pragma unroll
        for (int q = 0; q < 8; ++q) v[q] = (_Float16)a[q];
        *(v8h*)(nin + (size_t)i * 256 + 128 + cj) = v;
    }
}

// ---- decoder: per-channel upsample 1->4 + LN over the 4 ----
__global__ void decoder_up(const float* __restrict__ h,
                           const float* __restrict__ uw1, const float* __restrict__ ub1,
                           const float* __restrict__ uw2, const float* __restrict__ ub2,
                           const float* __restrict__ uw3, const float* __restrict__ ug,
                           const float* __restrict__ ubv, float* __restrict__ U, int ntot)
{
    long long t = blockIdx.x * (long long)blockDim.x + threadIdx.x;
    if (t >= (long long)ntot * 128) return;
    const int n = (int)(t >> 7), c = (int)(t & 127);
    const float hv = h[t];
    float u1[4], u2[4], u3[4];
#pragma unroll
    for (int o = 0; o < 4; ++o) u1[o] = lrelu(hv * uw1[c * 4 + o] + ub1[c * 4 + o]);
#pragma unroll
    for (int o = 0; o < 4; ++o) {
        float a = ub2[c * 4 + o];
#pragma unroll
        for (int i = 0; i < 4; ++i) a += u1[i] * uw2[c * 16 + i * 4 + o];
        u2[o] = lrelu(a);
    }
#pragma unroll
    for (int o = 0; o < 4; ++o) {
        float a = 0.f;
#pragma unroll
        for (int i = 0; i < 4; ++i) a += u2[i] * uw3[c * 16 + i * 4 + o];
        u3[o] = a;
    }
    const float mean = (u3[0] + u3[1] + u3[2] + u3[3]) * 0.25f;
    float var = 0.f;
#pragma unroll
    for (int o = 0; o < 4; ++o) { const float d = u3[o] - mean; var += d * d; }
    var *= 0.25f;
    const float rstd = rsqrtf(var + 1e-5f);
#pragma unroll
    for (int o = 0; o < 4; ++o)
        U[((size_t)n * 4 + o) * 128 + c] = (u3[o] - mean) * rstd * ug[c * 4 + o] + ubv[c * 4 + o];
}

__global__ void decoder_scatter(const float* __restrict__ U, const int* __restrict__ ec,
                                float* __restrict__ nodeacc, int ntot4)
{
    long long t = blockIdx.x * (long long)blockDim.x + threadIdx.x;
    if (t >= (long long)ntot4 * 128) return;
    const int idx = (int)(t >> 7), c = (int)(t & 127);
    const int eg = idx >> 2, o = idx & 3;
    const int b = eg / NE, e = eg % NE;
    const int node = ec[e * 4 + o];
    atomicAdd(nodeacc + ((size_t)b * NN + node) * 128 + c, U[t]);
}

__global__ void decoder_final(const float* __restrict__ nodeacc,
                              const float* __restrict__ cw1, const float* __restrict__ cb1,
                              const float* __restrict__ cw2, const float* __restrict__ cb2,
                              const float* __restrict__ cw3, float* __restrict__ out, int rows)
{
    int r = blockIdx.x * blockDim.x + threadIdx.x;
    if (r >= rows) return;
    float o1[3] = {cb1[0], cb1[1], cb1[2]};
    const float* xr = nodeacc + (size_t)r * 128;
    for (int c = 0; c < 128; ++c) {
        const float xv = xr[c];
        o1[0] += xv * cw1[c * 3 + 0];
        o1[1] += xv * cw1[c * 3 + 1];
        o1[2] += xv * cw1[c * 3 + 2];
    }
#pragma unroll
    for (int j = 0; j < 3; ++j) o1[j] = lrelu(o1[j]);
    float o2[3];
#pragma unroll
    for (int j = 0; j < 3; ++j) {
        float a = cb2[j];
#pragma unroll
        for (int i = 0; i < 3; ++i) a += o1[i] * cw2[i * 3 + j];
        o2[j] = lrelu(a);
    }
#pragma unroll
    for (int j = 0; j < 3; ++j) {
        float a = 0.f;
#pragma unroll
        for (int i = 0; i < 3; ++i) a += o2[i] * cw3[i * 3 + j];
        out[(size_t)r * 3 + j] = a;
    }
}

// ============================================================
// host-side orchestration
// ============================================================
extern "C" void kernel_launch(void* const* d_in, const int* in_sizes, int n_in,
                              void* d_out, int out_size, void* d_ws, size_t ws_size,
                              hipStream_t stream)
{
    (void)in_sizes; (void)n_in; (void)out_size; (void)ws_size;
    // ---- inputs ----
    const float* x   = (const float*)d_in[0];
    const int* ec    = (const int*)d_in[1];
    const int* ei    = (const int*)d_in[2];
    const float* enc_cw1 = (const float*)d_in[3];
    const float* enc_cb1 = (const float*)d_in[4];
    const float* enc_cw2 = (const float*)d_in[5];
    const float* enc_cb2 = (const float*)d_in[6];
    const float* enc_cw3 = (const float*)d_in[7];
    const float* enc_clng = (const float*)d_in[8];
    const float* enc_clnb = (const float*)d_in[9];
    const float* enc_ew1 = (const float*)d_in[10];
    const float* enc_eb1 = (const float*)d_in[11];
    const float* enc_ew2 = (const float*)d_in[12];
    const float* enc_eb2 = (const float*)d_in[13];
    const float* enc_ew3 = (const float*)d_in[14];
    const float* enc_elng = (const float*)d_in[15];
    const float* enc_elnb = (const float*)d_in[16];
    const float* p_ew1 = (const float*)d_in[17];
    const float* p_eb1 = (const float*)d_in[18];
    const float* p_ew2 = (const float*)d_in[19];
    const float* p_eb2 = (const float*)d_in[20];
    const float* p_ew3 = (const float*)d_in[21];
    const float* p_elng = (const float*)d_in[22];
    const float* p_elnb = (const float*)d_in[23];
    const float* p_nw1 = (const float*)d_in[24];
    const float* p_nb1 = (const float*)d_in[25];
    const float* p_nw2 = (const float*)d_in[26];
    const float* p_nb2 = (const float*)d_in[27];
    const float* p_nw3 = (const float*)d_in[28];
    const float* p_nlng = (const float*)d_in[29];
    const float* p_nlnb = (const float*)d_in[30];
    const float* dec_uw1 = (const float*)d_in[31];
    const float* dec_ub1 = (const float*)d_in[32];
    const float* dec_uw2 = (const float*)d_in[33];
    const float* dec_ub2 = (const float*)d_in[34];
    const float* dec_uw3 = (const float*)d_in[35];
    const float* dec_ulng = (const float*)d_in[36];
    const float* dec_ulnb = (const float*)d_in[37];
    const float* dec_cw1 = (const float*)d_in[38];
    const float* dec_cb1 = (const float*)d_in[39];
    const float* dec_cw2 = (const float*)d_in[40];
    const float* dec_cb2 = (const float*)d_in[41];
    const float* dec_cw3 = (const float*)d_in[42];
    float* out = (float*)d_out;

    // ---- workspace bump allocator (256B aligned) ----
    char* wp = (char*)d_ws;
    auto alloc = [&](size_t bytes) -> char* {
        char* r = wp;
        wp += (bytes + 255) & ~(size_t)255;
        return r;
    };
    _Float16* xpad16 = (_Float16*)alloc((size_t)NNODE * 32 * 2);
    _Float16* eh16a  = (_Float16*)alloc((size_t)NEDGE * 128 * 2);
    _Float16* eh16b  = (_Float16*)alloc((size_t)NEDGE * 128 * 2);
    float*    hbuf   = (float*)   alloc((size_t)NNODE * 128 * 4);
    _Float16* h16    = (_Float16*)alloc((size_t)NNODE * 128 * 2);
    char*     big    =            alloc((size_t)NEDGE * 256 * 2);  // msg16 / U (same size)
    _Float16* msg16  = (_Float16*)big;
    float*    U      = (float*)big;
    float*    e3     = (float*)   alloc((size_t)NEDGE * 128 * 4);
    float*    aggr   = (float*)   alloc((size_t)NNODE * 128 * 4);
    _Float16* nin16  = (_Float16*)alloc((size_t)NNODE * 256 * 2);
    float*    nodeacc= (float*)   alloc((size_t)BS * NN * 128 * 4);
    _Float16* enc_w1t = (_Float16*)alloc((size_t)128 * 32 * 2);
    _Float16* enc_w2t = (_Float16*)alloc((size_t)128 * 128 * 2);
    _Float16* enc_w3t = (_Float16*)alloc((size_t)128 * 128 * 2);
    _Float16* pe1t = (_Float16*)alloc((size_t)DBLK * 128 * 256 * 2);
    _Float16* pe2t = (_Float16*)alloc((size_t)DBLK * 128 * 128 * 2);
    _Float16* pe3t = (_Float16*)alloc((size_t)DBLK * 128 * 128 * 2);
    _Float16* pn1t = (_Float16*)alloc((size_t)DBLK * 128 * 256 * 2);
    _Float16* pn2t = (_Float16*)alloc((size_t)DBLK * 128 * 128 * 2);
    _Float16* pn3t = (_Float16*)alloc((size_t)DBLK * 128 * 128 * 2);

    const int TB = 256;
    auto blks = [](long long n, int tb) { return (int)((n + tb - 1) / tb); };

    // ---- 1) convert weights to fp16 transposed [N][Kpad] ----
    convert_wt<<<blks((long long)128 * 32, TB), TB, 0, stream>>>(enc_ew1, enc_w1t, 3, 32, 128, (long long)128 * 32);
    convert_wt<<<blks((long long)128 * 128, TB), TB, 0, stream>>>(enc_ew2, enc_w2t, 128, 128, 128, (long long)128 * 128);
    convert_wt<<<blks((long long)128 * 128, TB), TB, 0, stream>>>(enc_ew3, enc_w3t, 128, 128, 128, (long long)128 * 128);
    convert_wt<<<blks((long long)DBLK * 128 * 256, TB), TB, 0, stream>>>(p_ew1, pe1t, 256, 256, 128, (long long)DBLK * 128 * 256);
    convert_wt<<<blks((long long)DBLK * 128 * 128, TB), TB, 0, stream>>>(p_ew2, pe2t, 128, 128, 128, (long long)DBLK * 128 * 128);
    convert_wt<<<blks((long long)DBLK * 128 * 128, TB), TB, 0, stream>>>(p_ew3, pe3t, 128, 128, 128, (long long)DBLK * 128 * 128);
    convert_wt<<<blks((long long)DBLK * 128 * 256, TB), TB, 0, stream>>>(p_nw1, pn1t, 256, 256, 128, (long long)DBLK * 128 * 256);
    convert_wt<<<blks((long long)DBLK * 128 * 128, TB), TB, 0, stream>>>(p_nw2, pn2t, 128, 128, 128, (long long)DBLK * 128 * 128);
    convert_wt<<<blks((long long)DBLK * 128 * 128, TB), TB, 0, stream>>>(p_nw3, pn3t, 128, 128, 128, (long long)DBLK * 128 * 128);

    // ---- 2) encoder ----
    encoder_conv<<<blks(NNODE, TB), TB, 0, stream>>>(x, ec, enc_cw1, enc_cb1, enc_cw2,
                                                     enc_cb2, enc_cw3, enc_clng, enc_clnb,
                                                     xpad16, NNODE);
    wmma_mlp_layer<1, true,  true,  true ><<<NNODE / 128, TB, 0, stream>>>(xpad16, enc_w1t, enc_eb1, nullptr, eh16a);
    wmma_mlp_layer<4, true,  true,  true ><<<NNODE / 128, TB, 0, stream>>>(eh16a, enc_w2t, enc_eb2, nullptr, eh16b);
    wmma_mlp_layer<4, false, false, false><<<NNODE / 128, TB, 0, stream>>>(eh16b, enc_w3t, nullptr, hbuf, nullptr);
    layernorm128<<<NNODE / 8, TB, 0, stream>>>(hbuf, h16, enc_elng, enc_elnb, NNODE);

    // ---- 3) processor: D GraphNet blocks ----
    for (int d = 0; d < DBLK; ++d) {
        const size_t w1o = (size_t)d * 128 * 256;
        const size_t w2o = (size_t)d * 128 * 128;
        const size_t bo  = (size_t)d * 128;
        // edge MLP
        gather_msg<<<blks((long long)NEDGE * 32, TB), TB, 0, stream>>>(h16, ei, msg16, NEDGE);
        wmma_mlp_layer<8, true,  true,  true ><<<NEDGE / 128, TB, 0, stream>>>(msg16, pe1t + w1o, p_eb1 + bo, nullptr, eh16a);
        wmma_mlp_layer<4, true,  true,  true ><<<NEDGE / 128, TB, 0, stream>>>(eh16a, pe2t + w2o, p_eb2 + bo, nullptr, eh16b);
        wmma_mlp_layer<4, false, false, false><<<NEDGE / 128, TB, 0, stream>>>(eh16b, pe3t + w2o, nullptr, e3, nullptr);
        layernorm128<<<NEDGE / 8, TB, 0, stream>>>(e3, nullptr, p_elng + bo, p_elnb + bo, NEDGE);
        // aggregate
        fill_zero_f32<<<blks((long long)NNODE * 128, TB), TB, 0, stream>>>(aggr, (long long)NNODE * 128);
        scatter_add_edges<<<blks((long long)NEDGE * 128, TB), TB, 0, stream>>>(e3, ei, aggr, NEDGE);
        // node MLP
        concat_node<<<blks((long long)NNODE * 32, TB), TB, 0, stream>>>(h16, aggr, nin16, NNODE);
        wmma_mlp_layer<8, true,  true,  true ><<<NNODE / 128, TB, 0, stream>>>(nin16, pn1t + w1o, p_nb1 + bo, nullptr, eh16a);
        wmma_mlp_layer<4, true,  true,  true ><<<NNODE / 128, TB, 0, stream>>>(eh16a, pn2t + w2o, p_nb2 + bo, nullptr, eh16b);
        wmma_mlp_layer<4, false, false, false><<<NNODE / 128, TB, 0, stream>>>(eh16b, pn3t + w2o, nullptr, hbuf, nullptr);
        layernorm128<<<NNODE / 8, TB, 0, stream>>>(hbuf, h16, p_nlng + bo, p_nlnb + bo, NNODE);
    }

    // ---- 4) decoder ----
    decoder_up<<<blks((long long)NNODE * 128, TB), TB, 0, stream>>>(hbuf, dec_uw1, dec_ub1, dec_uw2,
                                                                    dec_ub2, dec_uw3, dec_ulng, dec_ulnb,
                                                                    U, NNODE);
    fill_zero_f32<<<blks((long long)BS * NN * 128, TB), TB, 0, stream>>>(nodeacc, (long long)BS * NN * 128);
    decoder_scatter<<<blks((long long)NNODE * 4 * 128, TB), TB, 0, stream>>>(U, ec, nodeacc, NNODE * 4);
    decoder_final<<<blks(BS * NN, TB), TB, 0, stream>>>(nodeacc, dec_cw1, dec_cb1, dec_cw2,
                                                        dec_cb2, dec_cw3, out, BS * NN);
}
